// LSTM_62723702391101
// MI455X (gfx1250) — matrix operand order repaired and checked by reading, not compiled, not run on hardware
//
#include <hip/hip_runtime.h>

#define BB 64
#define TT 512
#define DD 512
#define HH 512

// recurrence kernel geometry
#define RBLK 16          // persistent blocks
#define NSL  32          // N columns per block
#define URS  (DD + 8)    // padded LDS row stride (elements) -> 1040B rows, conflict-free b128
#define LDS_BYTES (4 * NSL * URS * 2)   // 133,120 B  (<= 320KB WGP LDS)

typedef __attribute__((ext_vector_type(16))) __bf16 v16bf;
typedef __attribute__((ext_vector_type(8)))  __bf16 v8bf;
typedef __attribute__((ext_vector_type(4)))  __bf16 v4bf;
typedef __attribute__((ext_vector_type(8)))  float  v8f;

#define WMMA_BF16(a, b, c) \
  __builtin_amdgcn_wmma_f32_16x16x32_bf16(false, (a), false, (b), (short)0, (c), false, false)

// ---- WMMA fragment loaders (wave32 layouts per CDNA5 ISA 7.12.2) ----
// A 16x32 bf16: lane half selects +8 K offset; elems 0..7 -> K base..base+7,
// elems 8..15 -> K base+16..base+23.
__device__ __forceinline__ v16bf frag_a(const __bf16* row, int k0, int lane) {
  const __bf16* p = row + k0 + ((lane & 16) ? 8 : 0);
  v8bf lo = *(const v8bf*)(p);
  v8bf hi = *(const v8bf*)(p + 16);
  v16bf r;
#pragma unroll
  for (int i = 0; i < 8; ++i) { r[i] = lo[i]; r[i + 8] = hi[i]; }
  return r;
}
// B 32x16 bf16 (row n contiguous in pre-transposed [N][K] storage):
// lane half selects +8 K offset; elems 0..7 -> K base..base+7, elems 8..15 -> K base+16..base+23.
__device__ __forceinline__ v16bf frag_b(const __bf16* row, int k0, int lane) {
  const __bf16* p = row + k0 + ((lane & 16) ? 8 : 0);
  v8bf lo = *(const v8bf*)(p);
  v8bf hi = *(const v8bf*)(p + 16);
  v16bf r;
#pragma unroll
  for (int i = 0; i < 8; ++i) { r[i] = lo[i]; r[i + 8] = hi[i]; }
  return r;
}

// ---- fp32 -> bf16 conversion of x ----
__global__ void lstm_cvt_x(const float* __restrict__ x, __bf16* __restrict__ xb) {
  int i = blockIdx.x * blockDim.x + threadIdx.x;     // over BB*TT*DD/4
  float4 v = ((const float4*)x)[i];
  v4bf o;
  o[0] = (__bf16)v.x; o[1] = (__bf16)v.y; o[2] = (__bf16)v.z; o[3] = (__bf16)v.w;
  ((v4bf*)xb)[i] = o;
}

// ---- W/U: fp32 [K][N] -> bf16 transposed [mat][N][K] ----
__global__ void lstm_cvt_wu(const float* w0, const float* w1, const float* w2, const float* w3,
                            const float* u0, const float* u1, const float* u2, const float* u3,
                            __bf16* __restrict__ dst) {
  int idx = blockIdx.x * blockDim.x + threadIdx.x;   // over 8*512*512
  int mat = idx >> 18;
  int rem = idx & 262143;
  int nn  = rem >> 9;
  int kk  = rem & 511;
  const float* src;
  switch (mat) {
    case 0: src = w0; break; case 1: src = w1; break;
    case 2: src = w2; break; case 3: src = w3; break;
    case 4: src = u0; break; case 5: src = u1; break;
    case 6: src = u2; break; default: src = u3; break;
  }
  dst[idx] = (__bf16)src[kk * HH + nn];
}

// ---- input-projection GEMM: P[g][t][b][n] = x @ W_g + bias_g ----
// 32x64 output block per wave (2 M-tiles x 4 N-tiles): 12 b128 loads per 8 WMMAs,
// independent fragment registers so loads clause and overlap the WMMAs.
__global__ void __launch_bounds__(256) lstm_gemm_x(
    const __bf16* __restrict__ xb,   // [BB*TT][DD]
    const __bf16* __restrict__ Wb,   // [4][HH][DD] (N-major)
    const float* __restrict__ bc, const float* __restrict__ bi,
    const float* __restrict__ bf_, const float* __restrict__ bo,
    float* __restrict__ P) {
  int lane = threadIdx.x & 31;
  int wid  = blockIdx.x * 8 + (threadIdx.x >> 5);
  int g  = wid >> 13;          // 4 gates x 8192 waves
  int r  = wid & 8191;
  int m0 = (r >> 3) << 5;      // 1024 M-pairs (32 rows each)
  int n0 = (r & 7) << 6;       // 8 N-strips of 64
  const __bf16* Wg = Wb + (size_t)g * HH * DD;

  v8f zero = {0.f,0.f,0.f,0.f,0.f,0.f,0.f,0.f};
  v8f acc0[4] = {zero, zero, zero, zero};
  v8f acc1[4] = {zero, zero, zero, zero};
  const __bf16* arow0 = xb + (size_t)(m0 + (lane & 15)) * DD;
  const __bf16* arow1 = arow0 + (size_t)16 * DD;
  const __bf16* brow0 = Wg + (size_t)(n0 + (lane & 15)) * DD;

  for (int k0 = 0; k0 < DD; k0 += 32) {
    v16bf a0 = frag_a(arow0, k0, lane);
    v16bf a1 = frag_a(arow1, k0, lane);
    v16bf b0 = frag_b(brow0,           k0, lane);
    v16bf b1 = frag_b(brow0 + 16 * DD, k0, lane);
    v16bf b2 = frag_b(brow0 + 32 * DD, k0, lane);
    v16bf b3 = frag_b(brow0 + 48 * DD, k0, lane);
    acc0[0] = WMMA_BF16(a0, b0, acc0[0]);
    acc0[1] = WMMA_BF16(a0, b1, acc0[1]);
    acc0[2] = WMMA_BF16(a0, b2, acc0[2]);
    acc0[3] = WMMA_BF16(a0, b3, acc0[3]);
    acc1[0] = WMMA_BF16(a1, b0, acc1[0]);
    acc1[1] = WMMA_BF16(a1, b1, acc1[1]);
    acc1[2] = WMMA_BF16(a1, b2, acc1[2]);
    acc1[3] = WMMA_BF16(a1, b3, acc1[3]);
  }
  const float* bias = (g == 0) ? bc : (g == 1) ? bi : (g == 2) ? bf_ : bo;
  int nl = lane & 15;
  int mh = (lane >> 4) << 3;
#pragma unroll
  for (int j = 0; j < 4; ++j) {
    float bv = bias[n0 + j * 16 + nl];
#pragma unroll
    for (int rr = 0; rr < 8; ++rr) {
      int m = m0 + mh + rr;              // m = b*T + t
      P[(((size_t)g * TT + (m & 511)) * BB + (m >> 9)) * HH + n0 + j * 16 + nl] =
          acc0[j][rr] + bv;
      m += 16;
      P[(((size_t)g * TT + (m & 511)) * BB + (m >> 9)) * HH + n0 + j * 16 + nl] =
          acc1[j][rr] + bv;
    }
  }
}

__global__ void lstm_init_sync(unsigned* s) { if (threadIdx.x < 64) s[threadIdx.x] = 0u; }

// monotonic-counter grid barrier (RBLK co-resident blocks)
__device__ __forceinline__ void grid_barrier(unsigned* cnt, unsigned target) {
  __syncthreads();
  if (threadIdx.x == 0) {
    __threadfence();
    __hip_atomic_fetch_add(cnt, 1u, __ATOMIC_ACQ_REL, __HIP_MEMORY_SCOPE_AGENT);
    while (__hip_atomic_load(cnt, __ATOMIC_ACQUIRE, __HIP_MEMORY_SCOPE_AGENT) < target) {
      __builtin_amdgcn_s_sleep(1);
    }
  }
  __syncthreads();
}

// ---- persistent recurrence: block j owns all 4 gates for N in [32j,32j+32) ----
// Per step: hoist full A(h) tile into registers (one load clause), then issue the
// 32 HBM P-loads so they complete under the 64-WMMA GEMM, then activations.
__global__ void __launch_bounds__(256) lstm_rec(
    const __bf16* __restrict__ Ub,   // [4][HH][HH] (N-major)
    const float* __restrict__ P,     // [4][TT][BB][HH]
    __bf16* __restrict__ hb,         // [2][BB][HH] double-buffered bf16 h
    float* __restrict__ out,         // [BB][TT][HH]
    unsigned* __restrict__ sync) {
  extern __shared__ __bf16 Us[];     // [4][NSL][URS]
  int lane = threadIdx.x & 31;
  int w    = threadIdx.x >> 5;       // 0..7
  int n0b  = blockIdx.x * NSL;

  // stage this block's U columns (4 gates x 32 rows x 512 K) into LDS, padded rows
  for (int c = threadIdx.x; c < 4 * NSL * (DD / 8); c += blockDim.x) {
    int g   = c >> 11;               // / (NSL*64)
    int rem = c & 2047;
    int nl  = rem >> 6;
    int kc  = rem & 63;
    const v8bf* src = (const v8bf*)(Ub + ((size_t)g * HH + n0b + nl) * DD + kc * 8);
    *(v8bf*)(&Us[(g * NSL + nl) * URS + kc * 8]) = *src;
  }
  // zero h buffer 0 for this block's slice
  for (int c = threadIdx.x; c < BB * NSL; c += blockDim.x) {
    int b = c >> 5, nl = c & 31;
    hb[b * HH + n0b + nl] = (__bf16)0.0f;
  }
  __syncthreads();
  unsigned target = RBLK;
  grid_barrier(sync, target);
  target += RBLK;

  int m0  = (w & 3) << 4;            // batch tile: 0,16,32,48
  int nl0 = (w >> 2) << 4;           // local n sub-tile: 0 or 16
  int n   = n0b + nl0 + (lane & 15); // this lane's output column
  int mh  = (lane >> 4) << 3;
  const __bf16* ldsrow = &Us[(nl0 + (lane & 15)) * URS];
  const size_t GSZ = (size_t)TT * BB * HH;

  v8f zero = {0.f,0.f,0.f,0.f,0.f,0.f,0.f,0.f};
  v8f c_st = zero;                   // cell state lives in registers for all T steps

  for (int t = 0; t < TT; ++t) {
    const __bf16* h_rd = hb + (size_t)(t & 1) * BB * HH;
    __bf16* h_wr       = hb + (size_t)((t + 1) & 1) * BB * HH;
    const __bf16* arow = h_rd + (size_t)(m0 + (lane & 15)) * HH;

    // (1) whole A tile -> registers: 32 b128 loads issued together (L2-resident h)
    v16bf a[16];
#pragma unroll
    for (int kk = 0; kk < 16; ++kk) a[kk] = frag_a(arow, kk * 32, lane);

    // (2) issue this step's P loads now; in-order return means the GEMM below only
    //     waits on the A group, and these HBM loads land during the 64 WMMAs
    size_t ptb = (size_t)t * BB * HH;
    float pld[4][8];
#pragma unroll
    for (int g = 0; g < 4; ++g)
#pragma unroll
      for (int rr = 0; rr < 8; ++rr)
        pld[g][rr] = P[g * GSZ + ptb + (size_t)(m0 + mh + rr) * HH + n];

    // (3) GEMM: 16 k-steps x 4 gates, B fragments from LDS
    v8f acc[4] = {zero, zero, zero, zero};
#pragma unroll
    for (int kk = 0; kk < 16; ++kk) {
      int k0 = kk * 32;
      v16bf b0 = frag_b(ldsrow + 0 * NSL * URS, k0, lane);
      v16bf b1 = frag_b(ldsrow + 1 * NSL * URS, k0, lane);
      v16bf b2 = frag_b(ldsrow + 2 * NSL * URS, k0, lane);
      v16bf b3 = frag_b(ldsrow + 3 * NSL * URS, k0, lane);
      acc[0] = WMMA_BF16(a[kk], b0, acc[0]);
      acc[1] = WMMA_BF16(a[kk], b1, acc[1]);
      acc[2] = WMMA_BF16(a[kk], b2, acc[2]);
      acc[3] = WMMA_BF16(a[kk], b3, acc[3]);
    }

    // (4) gates + state update, fp32
#pragma unroll
    for (int rr = 0; rr < 8; ++rr) {
      int b = m0 + mh + rr;
      float pc = pld[0][rr] + acc[0][rr];
      float pi = pld[1][rr] + acc[1][rr];
      float pf = pld[2][rr] + acc[2][rr];
      float po = pld[3][rr] + acc[3][rr];
      float av = tanhf(pc);
      float iv = 1.0f / (1.0f + __expf(-pi));
      float fv = 1.0f / (1.0f + __expf(-pf));
      float ov = 1.0f / (1.0f + __expf(-po));
      float cn = iv * av + fv * c_st[rr];
      c_st[rr] = cn;
      float hv = ov * tanhf(cn);
      out[((size_t)b * TT + t) * HH + n] = hv;
      h_wr[(size_t)b * HH + n] = (__bf16)hv;
    }
    grid_barrier(sync, target);      // h_wr visible device-wide before next step
    target += RBLK;
  }
}

extern "C" void kernel_launch(void* const* d_in, const int* in_sizes, int n_in,
                              void* d_out, int out_size, void* d_ws, size_t ws_size,
                              hipStream_t stream) {
  const float* x  = (const float*)d_in[0];
  const float* Wc = (const float*)d_in[1];
  const float* Wi = (const float*)d_in[2];
  const float* Wf = (const float*)d_in[3];
  const float* Wo = (const float*)d_in[4];
  const float* Uc = (const float*)d_in[5];
  const float* Ui = (const float*)d_in[6];
  const float* Uf = (const float*)d_in[7];
  const float* Uo = (const float*)d_in[8];
  const float* bc = (const float*)d_in[9];
  const float* bi = (const float*)d_in[10];
  const float* bf = (const float*)d_in[11];
  const float* bo = (const float*)d_in[12];
  float* out = (float*)d_out;

  // workspace carve-up (~306 MB)
  char* ws = (char*)d_ws;
  size_t off = 0;
  __bf16* xb  = (__bf16*)(ws + off); off += (size_t)BB * TT * DD * 2;     // 32 MB
  __bf16* WUb = (__bf16*)(ws + off); off += (size_t)8 * HH * DD * 2;      // 4 MB (W then U)
  float*  P   = (float*)(ws + off);  off += (size_t)4 * TT * BB * HH * 4; // 256 MB
  __bf16* hb  = (__bf16*)(ws + off); off += (size_t)2 * BB * HH * 2;      // 128 KB
  unsigned* sync = (unsigned*)(ws + off);

  lstm_cvt_x<<<(BB * TT * DD / 4) / 256, 256, 0, stream>>>(x, xb);
  lstm_cvt_wu<<<(8 * HH * DD) / 256, 256, 0, stream>>>(Wc, Wi, Wf, Wo, Uc, Ui, Uf, Uo, WUb);
  lstm_gemm_x<<<4096, 256, 0, stream>>>(xb, WUb, bc, bi, bf, bo, P);
  lstm_init_sync<<<1, 64, 0, stream>>>(sync);
  (void)hipFuncSetAttribute((const void*)lstm_rec,
                            hipFuncAttributeMaxDynamicSharedMemorySize, (int)LDS_BYTES);
  lstm_rec<<<RBLK, 256, LDS_BYTES, stream>>>(WUb + (size_t)4 * HH * DD, P, hb, out, sync);
}
